// PerceiverAttention_50680614092944
// MI455X (gfx1250) — compile-verified, matrix operand order checked by reading, not run on hardware
//
#include <hip/hip_runtime.h>
#include <hip/hip_bf16.h>

// ---------------------------------------------------------------------------
// Types for CDNA5 WMMA (wave32): 16x16x32 bf16 -> f32
// ---------------------------------------------------------------------------
typedef __attribute__((ext_vector_type(16))) __bf16 v16bf;
typedef __attribute__((ext_vector_type(8)))  float  v8f;
typedef __attribute__((ext_vector_type(4)))  float  vf4;   // 16-byte load/store unit

// TDM descriptor group types (amdgpu-toolchain 6-arg builtin signature)
typedef unsigned int u32x4 __attribute__((ext_vector_type(4)));
typedef int          i32x8 __attribute__((ext_vector_type(8)));
typedef int          i32x4 __attribute__((ext_vector_type(4)));

union FragB16 {
  v16bf  v;
  vf4    f4[2];
  __bf16 e[16];
};

__device__ __forceinline__ __bf16 f2bf(float f) {
  union { float f; unsigned u; } in; in.f = f;
  unsigned r = in.u + 0x7FFFu + ((in.u >> 16) & 1u);  // round-to-nearest-even
  union { unsigned short s; __bf16 b; } out;
  out.s = (unsigned short)(r >> 16);
  return out.b;
}

// ---------------------------------------------------------------------------
// TDM: DMA a bf16 tile (tile_rows x 64 elems, optionally 'chunks' row-groups
// separated by chunk_stride elems) from global into LDS.
// LDS pad: 16 bytes after every 128 bytes -> row stride 144 B = 72 bf16,
// matching the padded LDS layout the WMMA fragment reads expect.
// D# bitfields per CDNA5 ISA section 8.3/8.4.
// ---------------------------------------------------------------------------
__device__ __forceinline__ void tdm_load_tile(
    unsigned lds_addr, unsigned long gaddr,
    int tile_rows, int chunks, long chunk_stride_elems)
{
  u32x4 g0;
  g0[0] = 1u;                                            // count=1, user D#
  g0[1] = lds_addr;                                      // lds_addr (bytes)
  g0[2] = (unsigned)(gaddr & 0xFFFFFFFFu);               // global_addr[31:0]
  g0[3] = (unsigned)((gaddr >> 32) & 0x1FFFFFFu)         // global_addr[56:32]
        | (2u << 30);                                    // type=2 ("image")

  const unsigned td0 = 1024u;                            // tensor_dim0
  const unsigned td1 = 0x7FFFFFFFu;                      // tensor_dim1 (no OOB)
  const unsigned long s0 = 1024ul;                       // dim0 stride (elems)
  const unsigned long s1 = (unsigned long)chunk_stride_elems;

  i32x8 g1;
  // [17:16] data_size=1 (2B)  [20] pad_enable  [24:22] pad_interval=4 (32 dw)
  // [31:25] pad_amount=3 (4 dwords = 16 B)
  g1[0] = (1 << 16) | (1 << 20) | (4 << 22) | (3 << 25);
  g1[1] = (int)((td0 & 0xFFFFu) << 16);                  // [63:48] td0 lo
  g1[2] = (int)((td0 >> 16) | ((td1 & 0xFFFFu) << 16));  // td0 hi | td1 lo
  g1[3] = (int)((td1 >> 16) | (64u << 16));              // td1 hi | tile_dim0=64
  g1[4] = (int)((unsigned)tile_rows | ((unsigned)chunks << 16)); // tile_dim1|2
  g1[5] = (int)(s0 & 0xFFFFFFFFu);                       // dim0_stride[31:0]
  g1[6] = (int)(((s0 >> 32) & 0xFFFFu) | ((s1 & 0xFFFFu) << 16));
  g1[7] = (int)(s1 >> 16);                               // dim1_stride[47:16]

  i32x4 g2;
  g2[0] = 0x7FFFFFFF;   // tensor_dim2: large (no OOB on chunk index)
  g2[1] = 0;            // tensor_dim3
  g2[2] = 0;            // tensor_dim2_stride (dim3 unused)
  g2[3] = 0;            // tile_dim3 = 0 (unused)
  i32x4 g3; g3[0] = 0; g3[1] = 0; g3[2] = 0; g3[3] = 0;
  i32x8 g4;             // extra group required by the 6-arg toolchain form
  g4[0] = 0; g4[1] = 0; g4[2] = 0; g4[3] = 0;
  g4[4] = 0; g4[5] = 0; g4[6] = 0; g4[7] = 0;

  __builtin_amdgcn_tensor_load_to_lds(g0, g1, g2, g3, g4, 0);
}

// ---------------------------------------------------------------------------
// Fused LayerNorm (x rows 0..4095, latent rows 4096..4159 per batch) -> bf16
// A layout: [b*4160 + row][1024], row-major. One block per row, 256 threads.
// ---------------------------------------------------------------------------
__global__ __launch_bounds__(256) void ln_k(
    const float* __restrict__ x, const float* __restrict__ lat,
    const float* __restrict__ g1, const float* __restrict__ b1,
    const float* __restrict__ g2, const float* __restrict__ b2,
    __bf16* __restrict__ A)
{
  __shared__ float red[256], red2[256];
  const int tid = threadIdx.x;
  const int row = blockIdx.x;
  const int b = row / 4160, r = row % 4160;
  const float* src; const float* g; const float* bb;
  if (r < 4096) { src = x   + ((long)b * 4096 + r) * 1024;        g = g1; bb = b1; }
  else          { src = lat + ((long)b * 64 + (r - 4096)) * 1024; g = g2; bb = b2; }

  float v[4], s = 0.f, s2 = 0.f;
  #pragma unroll
  for (int j = 0; j < 4; ++j) {
    v[j] = src[tid + j * 256];
    s += v[j]; s2 += v[j] * v[j];
  }
  red[tid] = s; red2[tid] = s2;
  __syncthreads();
  for (int off = 128; off > 0; off >>= 1) {
    if (tid < off) { red[tid] += red[tid + off]; red2[tid] += red2[tid + off]; }
    __syncthreads();
  }
  float mu   = red[0]  * (1.f / 1024.f);
  float var  = red2[0] * (1.f / 1024.f) - mu * mu;
  float rstd = rsqrtf(var + 1e-5f);

  __bf16* dst = A + (long)row * 1024;
  #pragma unroll
  for (int j = 0; j < 4; ++j) {
    int i = tid + j * 256;
    dst[i] = f2bf((v[j] - mu) * rstd * g[i] + bb[i]);
  }
}

// ---------------------------------------------------------------------------
// One-time fp32 -> bf16 weight transpose: W (1024 x N) -> Wt (N x 1024).
// ---------------------------------------------------------------------------
__global__ __launch_bounds__(256) void cvt_t_k(const float* __restrict__ s,
                                               __bf16* __restrict__ d, int N) {
  __shared__ __bf16 t[32][33];
  const int n0 = blockIdx.x * 32, k0 = blockIdx.y * 32;
  #pragma unroll
  for (int j = 0; j < 4; ++j) {
    int e = threadIdx.x + j * 256;
    int kk = e >> 5, nn = e & 31;
    t[kk][nn] = f2bf(s[(long)(k0 + kk) * N + n0 + nn]);
  }
  __syncthreads();
  #pragma unroll
  for (int j = 0; j < 4; ++j) {
    int e = threadIdx.x + j * 256;
    int nn = e >> 5, kk = e & 31;
    d[(long)(n0 + nn) * 1024 + k0 + kk] = t[kk][nn];
  }
}

// ---------------------------------------------------------------------------
// bf16 GEMM, K=1024, block tile 128x128, BK=64, double-buffered LDS.
// Tiles staged by the Tensor Data Mover: wave 0 issues TENSOR_LOAD_TO_LDS for
// the next tile (A and B) into the spare buffer while all 8 waves run WMMAs
// on the current one; wave 0 waits TENSORcnt==0 before the publishing barrier.
// mode 0: out = K/V head-major   mode 1: A-row remap (latents) + Q head-major
// mode 2: fp32 row-major out
// ---------------------------------------------------------------------------
__global__ __launch_bounds__(256) void gemm_bf16_k(
    const __bf16* __restrict__ A, const __bf16* __restrict__ Bt, int mode,
    __bf16* __restrict__ o0, __bf16* __restrict__ o1, float* __restrict__ of)
{
  __shared__ __bf16 As[2][128 * 72];   // 128 M rows x 64 K, TDM-padded to 72
  __shared__ __bf16 Bs[2][128 * 72];   // 128 N rows x 64 K, TDM-padded to 72

  const int tid  = threadIdx.x;
  const int wave = tid >> 5, lane = tid & 31;
  const int half = lane >> 4, lr = lane & 15;
  const int waveM = wave >> 2, waveN = wave & 3;
  const int bm = blockIdx.y * 128, bn = blockIdx.x * 128;

  v8f acc[4][2];
  #pragma unroll
  for (int m = 0; m < 4; ++m)
    #pragma unroll
    for (int n = 0; n < 2; ++n)
      #pragma unroll
      for (int i = 0; i < 8; ++i) acc[m][n][i] = 0.f;

  // tile geometry for the TDM descriptors
  const long arow0 = (mode == 1) ? ((long)(bm >> 6) * 4160 + 4096) : (long)bm;
  const __bf16* abase = A  + arow0 * 1024;
  const __bf16* bbase = Bt + (long)bn * 1024;
  const int  a_rows    = (mode == 1) ? 64 : 128;
  const int  a_chunks  = (mode == 1) ? 2 : 1;
  const long a_cstride = (long)4160 * 1024;   // elems between batch chunks

  auto issue_tiles = [&](int k0, int buf) {
    tdm_load_tile((unsigned)(size_t)(void*)&As[buf][0],
                  (unsigned long)(size_t)(abase + k0),
                  a_rows, a_chunks, a_cstride);
    tdm_load_tile((unsigned)(size_t)(void*)&Bs[buf][0],
                  (unsigned long)(size_t)(bbase + k0),
                  128, 1, 0);
  };

  auto compute = [&](int buf) {
    #pragma unroll
    for (int ks = 0; ks < 2; ++ks) {
      FragB16 af[4], bfr[2];
      #pragma unroll
      for (int mf = 0; mf < 4; ++mf) {
        int r = waveM * 64 + mf * 16 + lr;
        af[mf].f4[0] = *(const vf4*)&As[buf][r * 72 + ks * 32 + half * 8];
        af[mf].f4[1] = *(const vf4*)&As[buf][r * 72 + ks * 32 + 16 + half * 8];
      }
      #pragma unroll
      for (int nf = 0; nf < 2; ++nf) {
        int c = waveN * 32 + nf * 16 + lr;
        bfr[nf].f4[0] = *(const vf4*)&Bs[buf][c * 72 + ks * 32 + half * 8];
        bfr[nf].f4[1] = *(const vf4*)&Bs[buf][c * 72 + ks * 32 + 16 + half * 8];
      }
      #pragma unroll
      for (int mf = 0; mf < 4; ++mf)
        #pragma unroll
        for (int nf = 0; nf < 2; ++nf)
          acc[mf][nf] = __builtin_amdgcn_wmma_f32_16x16x32_bf16(
              false, af[mf].v, false, bfr[nf].v, (short)0, acc[mf][nf],
              false, false);
    }
  };

  if (wave == 0) {                       // prologue: tile 0 into buffer 0
    issue_tiles(0, 0);
    __builtin_amdgcn_s_wait_tensorcnt(0);
  }
  __syncthreads();
  for (int it = 0; it < 16; ++it) {
    const int cur = it & 1;
    if (it < 15 && wave == 0) issue_tiles((it + 1) * 64, cur ^ 1);
    compute(cur);
    if (wave == 0) __builtin_amdgcn_s_wait_tensorcnt(0);
    __syncthreads();                     // publishes next tile; guards reuse
  }

  // epilogue: C layout -> row = ...half*8+rr, col = ...+lane%16
  for (int mf = 0; mf < 4; ++mf) {
    for (int rr = 0; rr < 8; ++rr) {
      int row = bm + waveM * 64 + mf * 16 + half * 8 + rr;
      for (int nf = 0; nf < 2; ++nf) {
        int col = bn + waveN * 32 + nf * 16 + lr;
        float val = acc[mf][nf][rr];
        if (mode == 0) {            // split K/V, store head-major [bh][key][d]
          int b = row / 4160, key = row % 4160;
          int cc = col;
          if (cc < 1024) {
            int h = cc >> 6, d = cc & 63;
            o0[(((long)(b * 16 + h)) * 4160 + key) * 64 + d] = f2bf(val);
          } else {
            cc -= 1024;
            int h = cc >> 6, d = cc & 63;
            o1[(((long)(b * 16 + h)) * 4160 + key) * 64 + d] = f2bf(val);
          }
        } else if (mode == 1) {     // Q head-major [bh][q][d]
          int b = row >> 6, i = row & 63;
          int h = col >> 6, d = col & 63;
          o0[(((long)(b * 16 + h)) * 64 + i) * 64 + d] = f2bf(val);
        } else {                    // fp32 row-major (final output)
          of[(long)row * 1024 + col] = val;
        }
      }
    }
  }
}

// ---------------------------------------------------------------------------
// Flash-style attention: one block per (b,h), 4 waves x 16 queries,
// 65 tiles of 64 keys. scale^2 = 1/8 folded into logits.
// ---------------------------------------------------------------------------
__global__ __launch_bounds__(128) void attn_k(
    const __bf16* __restrict__ Q, const __bf16* __restrict__ K,
    const __bf16* __restrict__ V, const float* __restrict__ mask,
    __bf16* __restrict__ AO)
{
  __shared__ __bf16 Ks[64 * 72];   // [key][d], padded
  __shared__ __bf16 Vs[64 * 72];   // transposed [d][key], padded
  __shared__ __bf16 Ps[64 * 72];   // probabilities [q][key], padded

  const int tid = threadIdx.x;
  const int wave = tid >> 5, lane = tid & 31;
  const int half = lane >> 4, lr = lane & 15;
  const int bh = blockIdx.x;
  const int b = bh >> 4, h = bh & 15;

  FragB16 qf[2];
  {
    const __bf16* qbase = Q + (long)bh * 64 * 64 + (wave * 16 + lr) * 64;
    #pragma unroll
    for (int ks = 0; ks < 2; ++ks) {
      qf[ks].f4[0] = *(const vf4*)(qbase + ks * 32 + half * 8);
      qf[ks].f4[1] = *(const vf4*)(qbase + ks * 32 + 16 + half * 8);
    }
  }

  v8f accO[4];
  #pragma unroll
  for (int d = 0; d < 4; ++d)
    #pragma unroll
    for (int i = 0; i < 8; ++i) accO[d][i] = 0.f;
  float mrow[8], lrow[8];
  #pragma unroll
  for (int i = 0; i < 8; ++i) { mrow[i] = -1e30f; lrow[i] = 0.f; }

  const int ldrow = tid >> 1, ldseg = (tid & 1) * 32;
  const __bf16* Kbase = K + (long)bh * 4160 * 64;
  const __bf16* Vbase = V + (long)bh * 4160 * 64;

  for (int t = 0; t < 65; ++t) {
    const int kb = t * 64;
    { // K tile: 32 contiguous bf16 per thread
      const __bf16* src = Kbase + (long)(kb + ldrow) * 64 + ldseg;
      #pragma unroll
      for (int j = 0; j < 4; ++j)
        *(vf4*)&Ks[ldrow * 72 + ldseg + j * 8] = *(const vf4*)(src + j * 8);
    }
    { // V tile transposed into LDS
      union { vf4 f4[4]; __bf16 e[32]; } t2;
      const __bf16* src = Vbase + (long)(kb + ldrow) * 64 + ldseg;
      #pragma unroll
      for (int j = 0; j < 4; ++j) t2.f4[j] = *(const vf4*)(src + j * 8);
      #pragma unroll
      for (int j = 0; j < 32; ++j) Vs[(ldseg + j) * 72 + ldrow] = t2.e[j];
    }
    __syncthreads();

    v8f accS[4];
    #pragma unroll
    for (int n = 0; n < 4; ++n)
      #pragma unroll
      for (int i = 0; i < 8; ++i) accS[n][i] = 0.f;
    #pragma unroll
    for (int ks = 0; ks < 2; ++ks) {
      #pragma unroll
      for (int nf = 0; nf < 4; ++nf) {
        FragB16 kf;
        int krl = nf * 16 + lr;
        kf.f4[0] = *(const vf4*)&Ks[krl * 72 + ks * 32 + half * 8];
        kf.f4[1] = *(const vf4*)&Ks[krl * 72 + ks * 32 + 16 + half * 8];
        accS[nf] = __builtin_amdgcn_wmma_f32_16x16x32_bf16(
            false, qf[ks].v, false, kf.v, (short)0, accS[nf], false, false);
      }
    }

    float bias[4];
    #pragma unroll
    for (int nf = 0; nf < 4; ++nf) {
      int kg = kb + nf * 16 + lr;
      bias[nf] = (kg < 4096) ? (mask[(long)b * 4096 + kg] - 1.f) * 100.f : 0.f;
    }

    #pragma unroll
    for (int rr = 0; rr < 8; ++rr) {
      float sv[4], tm = -1e30f;
      #pragma unroll
      for (int nf = 0; nf < 4; ++nf) {
        sv[nf] = accS[nf][rr] * 0.125f + bias[nf];
        tm = fmaxf(tm, sv[nf]);
      }
      #pragma unroll
      for (int off = 1; off < 16; off <<= 1)
        tm = fmaxf(tm, __shfl_xor(tm, off, 32));
      float nm = fmaxf(mrow[rr], tm);
      float sc = __expf(mrow[rr] - nm);
      mrow[rr] = nm;
      float ps = 0.f;
      #pragma unroll
      for (int nf = 0; nf < 4; ++nf) {
        float p = __expf(sv[nf] - nm);
        ps += p;
        Ps[(wave * 16 + half * 8 + rr) * 72 + nf * 16 + lr] = f2bf(p);
      }
      #pragma unroll
      for (int off = 1; off < 16; off <<= 1)
        ps += __shfl_xor(ps, off, 32);
      lrow[rr] = lrow[rr] * sc + ps;
      #pragma unroll
      for (int df = 0; df < 4; ++df) accO[df][rr] *= sc;
    }
    __syncthreads();

    #pragma unroll
    for (int ks = 0; ks < 2; ++ks) {
      FragB16 pf;
      int prow = wave * 16 + lr;
      pf.f4[0] = *(const vf4*)&Ps[prow * 72 + ks * 32 + half * 8];
      pf.f4[1] = *(const vf4*)&Ps[prow * 72 + ks * 32 + 16 + half * 8];
      #pragma unroll
      for (int df = 0; df < 4; ++df) {
        FragB16 vfr;
        int drl = df * 16 + lr;
        vfr.f4[0] = *(const vf4*)&Vs[drl * 72 + ks * 32 + half * 8];
        vfr.f4[1] = *(const vf4*)&Vs[drl * 72 + ks * 32 + 16 + half * 8];
        accO[df] = __builtin_amdgcn_wmma_f32_16x16x32_bf16(
            false, pf.v, false, vfr.v, (short)0, accO[df], false, false);
      }
    }
    __syncthreads();
  }

  #pragma unroll
  for (int df = 0; df < 4; ++df)
    #pragma unroll
    for (int rr = 0; rr < 8; ++rr) {
      int q = wave * 16 + half * 8 + rr;
      float val = accO[df][rr] / lrow[rr];
      int row = b * 64 + q;
      int col = h * 64 + df * 16 + lr;
      AO[(long)row * 1024 + col] = f2bf(val);
    }
}

// ---------------------------------------------------------------------------
extern "C" void kernel_launch(void* const* d_in, const int* in_sizes, int n_in,
                              void* d_out, int out_size, void* d_ws, size_t ws_size,
                              hipStream_t stream) {
  (void)in_sizes; (void)n_in; (void)out_size; (void)ws_size;

  const float* x    = (const float*)d_in[0];
  const float* lat  = (const float*)d_in[1];
  const float* mask = (const float*)d_in[2];
  const float* g1   = (const float*)d_in[3];
  const float* b1   = (const float*)d_in[4];
  const float* g2   = (const float*)d_in[5];
  const float* b2   = (const float*)d_in[6];
  const float* Wq   = (const float*)d_in[7];
  const float* Wkv  = (const float*)d_in[8];
  const float* Wout = (const float*)d_in[9];

  // workspace carve (bf16 elements)
  __bf16* A      = (__bf16*)d_ws;                       // 33280*1024
  __bf16* Wq_t   = A      + (long)33280 * 1024;         // transposed 1024x1024
  __bf16* Wkv_t  = Wq_t   + (long)1024 * 1024;          // transposed 2048x1024
  __bf16* Wout_t = Wkv_t  + (long)1024 * 2048;          // transposed 1024x1024
  __bf16* Qb     = Wout_t + (long)1024 * 1024;          // [bh][64][64]
  __bf16* Kb     = Qb     + (long)128 * 64 * 64;        // [bh][4160][64]
  __bf16* Vb     = Kb     + (long)128 * 4160 * 64;
  __bf16* AO     = Vb     + (long)128 * 4160 * 64;      // [512][1024]

  ln_k<<<33280, 256, 0, stream>>>(x, lat, g1, b1, g2, b2, A);
  cvt_t_k<<<dim3(32, 32), 256, 0, stream>>>(Wq,   Wq_t,   1024);
  cvt_t_k<<<dim3(64, 32), 256, 0, stream>>>(Wkv,  Wkv_t,  2048);
  cvt_t_k<<<dim3(32, 32), 256, 0, stream>>>(Wout, Wout_t, 1024);

  // kv = A @ Wkv   (M=33280, N=2048) -> K/V head-major
  gemm_bf16_k<<<dim3(16, 260), 256, 0, stream>>>(A, Wkv_t, 0, Kb, Vb, nullptr);
  // q = ln @ Wq    (M=512 via latent-row remap, N=1024) -> Q head-major
  gemm_bf16_k<<<dim3(8, 4), 256, 0, stream>>>(A, Wq_t, 1, Qb, nullptr, nullptr);
  // attention
  attn_k<<<128, 128, 0, stream>>>(Qb, Kb, Vb, mask, AO);
  // out = AO @ Wout (M=512, N=1024) -> fp32 d_out
  gemm_bf16_k<<<dim3(8, 4), 256, 0, stream>>>(AO, Wout_t, 2, nullptr, nullptr,
                                              (float*)d_out);
}
